// WeightedModel_60327110639808
// MI455X (gfx1250) — compile-verified, matrix operand order checked by reading, not run on hardware
//
#include <hip/hip_runtime.h>

// Fused dead-graph-eliminated model:
//   out = relu(h @ W1[:64,:] + b1) @ W2[:64,:] + b2
// (edge weights are hardcoded zero in the reference, so all scatter terms are 0)
//
// f16 WMMA (v_wmma_f32_16x16x32_f16), f32 accumulate. Memory-bound design:
// one streaming pass over h (25.6MB) + out (12.8MB) at HBM; weights (24KB)
// stay in L2 and are held as register-resident B-fragments per wave.

typedef __attribute__((ext_vector_type(16))) _Float16 v16h;
typedef __attribute__((ext_vector_type(8)))  _Float16 v8h;
typedef __attribute__((ext_vector_type(8)))  float    v8f;
typedef __attribute__((ext_vector_type(4)))  float    v4f;

#define WAVES_PER_BLOCK 8

__global__ __launch_bounds__(256) void sage_as_mlp_wmma(
    const float* __restrict__ h,    // [N,64]  row-major
    const float* __restrict__ W1,   // [128,64] (only rows 0..63 reachable)
    const float* __restrict__ b1,   // [64]
    const float* __restrict__ W2,   // [128,32] (only rows 0..63 reachable)
    const float* __restrict__ b2,   // [32]
    float* __restrict__ out,        // [N,32]
    int N, int nTiles)
{
    // per-wave H1 staging tile, 16x64 f16 row-major (2KB/wave)
    __shared__ __align__(16) _Float16 h1s[WAVES_PER_BLOCK][16 * 64];

    const int lane    = threadIdx.x & 31;
    const int wave    = threadIdx.x >> 5;
    const int halfsel = lane >> 4;      // 0: lanes 0-15, 1: lanes 16-31
    const int l15     = lane & 15;
    const int kh      = halfsel * 8;    // K sub-offset within a 32-wide chunk

    // ---------------- register-resident weight B-fragments ----------------
    // B frag (K=32 x N=16), f16: lane holds column (lane&15);
    //   b[i]   = W[k0 + kh + i     ][col]   i = 0..7   (VGPR0-3)
    //   b[8+i] = W[k0 + kh + 16 + i][col]   i = 0..7   (VGPR4-7)
    v16h w1f[2][4];                     // [k-chunk][n-tile]
    #pragma unroll
    for (int kc = 0; kc < 2; ++kc) {
        #pragma unroll
        for (int nt = 0; nt < 4; ++nt) {
            const float* p = W1 + (size_t)(kc * 32 + kh) * 64 + nt * 16 + l15;
            v16h b;
            #pragma unroll
            for (int i = 0; i < 8; ++i) {
                b[i]     = (_Float16)p[(size_t)i * 64];
                b[8 + i] = (_Float16)p[(size_t)(16 + i) * 64];
            }
            w1f[kc][nt] = b;
        }
    }
    v16h w2f[2][2];
    #pragma unroll
    for (int kc = 0; kc < 2; ++kc) {
        #pragma unroll
        for (int nt = 0; nt < 2; ++nt) {
            const float* p = W2 + (size_t)(kc * 32 + kh) * 32 + nt * 16 + l15;
            v16h b;
            #pragma unroll
            for (int i = 0; i < 8; ++i) {
                b[i]     = (_Float16)p[(size_t)i * 32];
                b[8 + i] = (_Float16)p[(size_t)(16 + i) * 32];
            }
            w2f[kc][nt] = b;
        }
    }
    // biases: same column for all 8 rows held by this lane
    float b1v[4], b2v[2];
    #pragma unroll
    for (int nt = 0; nt < 4; ++nt) b1v[nt] = b1[nt * 16 + l15];
    #pragma unroll
    for (int nt = 0; nt < 2; ++nt) b2v[nt] = b2[nt * 16 + l15];

    _Float16* h1 = &h1s[wave][0];
    const int waveId = blockIdx.x * WAVES_PER_BLOCK + wave;
    const int nWaves = gridDim.x * WAVES_PER_BLOCK;

    for (int t = waveId; t < nTiles; t += nWaves) {
        // ---------------- GEMM1: X(16x64) @ W1(64x64) -> H1, +b1, relu ----
        v8f acc[4];
        #pragma unroll
        for (int nt = 0; nt < 4; ++nt) { v8f z = {}; acc[nt] = z; }

        #pragma unroll
        for (int kc = 0; kc < 2; ++kc) {
            // A frag: lane holds row (lane&15), K = kc*32 + kh + {0..7,16..23}
            int gr = t * 16 + l15;
            if (gr >= N) gr = N - 1;            // clamp (partial-tile safety)
            const float* xr = h + (size_t)gr * 64 + kc * 32 + kh;
            v4f x0 = *(const v4f*)(xr);
            v4f x1 = *(const v4f*)(xr + 4);
            v4f x2 = *(const v4f*)(xr + 16);
            v4f x3 = *(const v4f*)(xr + 20);
            v16h a;
            #pragma unroll
            for (int i = 0; i < 4; ++i) {
                a[i]      = (_Float16)x0[i];
                a[4 + i]  = (_Float16)x1[i];
                a[8 + i]  = (_Float16)x2[i];
                a[12 + i] = (_Float16)x3[i];
            }
            #pragma unroll
            for (int nt = 0; nt < 4; ++nt)
                acc[nt] = __builtin_amdgcn_wmma_f32_16x16x32_f16(
                    false, a, false, w1f[kc][nt], (short)0, acc[nt], false, false);
        }

        // bias + relu + stage to per-wave LDS (row-major 16x64 f16).
        // C layout: lanes 0-15 -> rows 0..7 (VGPR r = row r), lanes 16-31 -> rows 8..15.
        #pragma unroll
        for (int nt = 0; nt < 4; ++nt) {
            #pragma unroll
            for (int r = 0; r < 8; ++r) {
                float v = acc[nt][r] + b1v[nt];
                v = v > 0.0f ? v : 0.0f;
                const int row = halfsel * 8 + r;
                h1[row * 64 + nt * 16 + l15] = (_Float16)v;
            }
        }
        // Same-wave DS write->read dependency: ordered via DScnt (compiler-inserted
        // s_wait_dscnt). No workgroup barrier needed: buffer is private per wave.

        // ---------------- GEMM2: H1(16x64) @ W2(64x32) -> out, +b2 --------
        v8f acc2[2];
        #pragma unroll
        for (int nt = 0; nt < 2; ++nt) { v8f z = {}; acc2[nt] = z; }

        #pragma unroll
        for (int kc = 0; kc < 2; ++kc) {
            const _Float16* ar = h1 + l15 * 64 + kc * 32 + kh;
            v8h a0 = *(const v8h*)(ar);         // K = kh + 0..7   (ds_load_b128)
            v8h a1 = *(const v8h*)(ar + 16);    // K = kh + 16..23 (ds_load_b128)
            v16h a;
            #pragma unroll
            for (int i = 0; i < 8; ++i) { a[i] = a0[i]; a[8 + i] = a1[i]; }
            #pragma unroll
            for (int nt = 0; nt < 2; ++nt)
                acc2[nt] = __builtin_amdgcn_wmma_f32_16x16x32_f16(
                    false, a, false, w2f[kc][nt], (short)0, acc2[nt], false, false);
        }

        #pragma unroll
        for (int nt = 0; nt < 2; ++nt) {
            #pragma unroll
            for (int r = 0; r < 8; ++r) {
                const int row = t * 16 + halfsel * 8 + r;
                if (row < N)
                    out[(size_t)row * 32 + nt * 16 + l15] = acc2[nt][r] + b2v[nt];
            }
        }
    }
}

extern "C" void kernel_launch(void* const* d_in, const int* in_sizes, int n_in,
                              void* d_out, int out_size, void* d_ws, size_t ws_size,
                              hipStream_t stream) {
    const float* h  = (const float*)d_in[0];
    const float* W1 = (const float*)d_in[1];
    const float* b1 = (const float*)d_in[2];
    const float* W2 = (const float*)d_in[3];
    const float* b2 = (const float*)d_in[4];
    // d_in[5] = src, d_in[6] = dst: mathematically irrelevant (edge weights == 0
    // in the reference model => all neighbor aggregates are exactly zero).
    float* out = (float*)d_out;

    const int N      = in_sizes[0] / 64;       // 100000
    const int nTiles = (N + 15) / 16;          // 6250 (exact)

    int blocks = 256;                          // 2048 wave32s, grid-stride tiles
    if (blocks * WAVES_PER_BLOCK > nTiles)
        blocks = (nTiles + WAVES_PER_BLOCK - 1) / WAVES_PER_BLOCK;
    if (blocks < 1) blocks = 1;

    sage_as_mlp_wmma<<<blocks, 256, 0, stream>>>(h, W1, b1, W2, b2, out, N, nTiles);
}
